// ARNNDecoder_56178172231681
// MI455X (gfx1250) — matrix operand order repaired and checked by reading, not compiled, not run on hardware
//
#include <hip/hip_runtime.h>

#define B_      64
#define DEC_T_  128
#define ENC_T_  1024
#define H_      512
#define ENC_D_  512
#define ATTN_   512
#define VOCAB_  128
#define G3_     1536   // 3*H
#define RIN_    1024   // H + ENC_D

typedef __bf16 v16bf __attribute__((ext_vector_type(16)));
typedef __bf16 v8bf  __attribute__((ext_vector_type(8)));
typedef float  v8f   __attribute__((ext_vector_type(8)));

// Native converts (clang lowers to gfx1250 cvt ops, RTNE) instead of
// bit-twiddled versions that cost 4-5 VALU ops each.
static __device__ __forceinline__ __bf16 f2bf(float f) { return (__bf16)f; }
static __device__ __forceinline__ float bf2f(__bf16 b) { return (float)b; }

// gfx1250 native TRANS tanh. ISA 7.4: TRANS result needs 1 independent op
// before use -> bundle a v_nop to satisfy the hazard regardless of scheduling.
static __device__ __forceinline__ float fast_tanh(float x) {
  float r;
  asm("v_tanh_f32 %0, %1\n\tv_nop" : "=v"(r) : "v"(x));
  return r;
}
static __device__ __forceinline__ float sigmoidf_(float x) {
  return 1.0f / (1.0f + __expf(-x));   // native v_exp_f32 path
}

// ---------------------------------------------------------------------------
// Generic bf16 WMMA GEMM:  D(MxN) = A(MxK) * W(NxK)^T + bias
// One wave (32 threads) per 16x16 M-tile x 4 N-tiles. K stepped by 32.
// A layout: lane<16 row=lane, elems[0..15]=K[k..k+15];
//   lane>=16 same row, K offset +16 (matches B's contiguous k order).
// B = W^T: lane col n = lane&15; elems[0..15] = W[n][kb..kb+15], kb=k+(lane>>4)*16
//   -> 16 contiguous bf16 per lane (row-major weight gives transposed B for free).
// C/D: vgpr i -> row = i + (lane>>4)*8, col = lane&15.
// ---------------------------------------------------------------------------
template <bool A_IS_F32, bool D_IS_BF16>
__global__ __launch_bounds__(32) void gemm_bf16_wmma(
    const void* __restrict__ Ap, const __bf16* __restrict__ W,
    const float* __restrict__ bias, void* __restrict__ Dp,
    int M, int N, int K)
{
  const int lane = threadIdx.x;
  const int r    = lane & 15;
  const int hi   = lane >> 4;
  const int m0   = blockIdx.y * 16;
  const int n0   = blockIdx.x * 64;
  const size_t row = (size_t)(m0 + r);

  v8f acc[4] = {{}, {}, {}, {}};
  const __bf16* Ab = (const __bf16*)Ap;
  const float*  Af = (const float*)Ap;

  for (int k = 0; k < K; k += 32) {
    v16bf a;
    if (A_IS_F32) {
      const float* p0 = Af + row * (size_t)K + k + hi * 16;
#pragma unroll
      for (int j = 0; j < 8; ++j) a[j] = f2bf(p0[j]);
#pragma unroll
      for (int j = 0; j < 8; ++j) a[8 + j] = f2bf(p0[8 + j]);
    } else {
      const __bf16* p0 = Ab + row * (size_t)K + k + hi * 16;
      v8bf lo = *(const v8bf*)(p0);
      v8bf hh = *(const v8bf*)(p0 + 8);
#pragma unroll
      for (int j = 0; j < 8; ++j) { a[j] = lo[j]; a[8 + j] = hh[j]; }
    }
    const int kb = k + hi * 16;
#pragma unroll
    for (int jt = 0; jt < 4; ++jt) {
      const __bf16* wp = W + (size_t)(n0 + jt * 16 + r) * (size_t)K + kb;
      v16bf b = *(const v16bf*)wp;
      acc[jt] = __builtin_amdgcn_wmma_f32_16x16x32_bf16(
          false, a, false, b, (short)0, acc[jt], false, false);
    }
  }

#pragma unroll
  for (int jt = 0; jt < 4; ++jt) {
    const int col = n0 + jt * 16 + r;
    const float bv = bias ? bias[col] : 0.0f;
#pragma unroll
    for (int i = 0; i < 8; ++i) {
      const size_t orow = (size_t)(m0 + i + hi * 8);
      const float v = acc[jt][i] + bv;
      if (D_IS_BF16) ((__bf16*)Dp)[orow * (size_t)N + col] = f2bf(v);
      else           ((float*)Dp)[orow * (size_t)N + col]  = v;
    }
  }
}

// ---------------------------------------------------------------------------
// One-time helpers
// ---------------------------------------------------------------------------
__global__ void cvt_f32_bf16(const float* __restrict__ in,
                             __bf16* __restrict__ out, int n) {
  int i = blockIdx.x * 256 + threadIdx.x;
  if (i < n) out[i] = f2bf(in[i]);
}

__global__ void bias_sum_k(const float* __restrict__ a,
                           const float* __restrict__ b,
                           float* __restrict__ out, int n) {
  int i = blockIdx.x * 256 + threadIdx.x;
  if (i < n) out[i] = a[i] + b[i];
}

__global__ void embed_k(const int* __restrict__ inputs,
                        const float* __restrict__ embedding,
                        __bf16* __restrict__ emb_bf) {
  int idx = blockIdx.x * 256 + threadIdx.x;   // B*DEC_T*H threads
  int j = idx & (H_ - 1);
  int bt = idx >> 9;
  int tok = inputs[bt];
  emb_bf[idx] = f2bf(embedding[(size_t)tok * H_ + j]);
}

// ---------------------------------------------------------------------------
// Per-step kernels
// ---------------------------------------------------------------------------
__global__ void pack_rnn_in_k(const __bf16* __restrict__ emb_bf,
                              const float* __restrict__ ctx,
                              __bf16* __restrict__ rnn_in, int t) {
  int idx = blockIdx.x * 256 + threadIdx.x;  // B*RIN
  int b = idx >> 10, j = idx & (RIN_ - 1);
  __bf16 v;
  if (j < H_) v = emb_bf[((size_t)(b * DEC_T_ + t)) * H_ + j];
  else        v = f2bf(ctx[b * ENC_D_ + (j - H_)]);
  rnn_in[idx] = v;
}

__global__ void gru_update_k(const float* __restrict__ gi,
                             const float* __restrict__ gh,
                             float* __restrict__ h,
                             __bf16* __restrict__ h_bf) {
  int idx = blockIdx.x * 256 + threadIdx.x;  // B*H
  int b = idx >> 9, j = idx & (H_ - 1);
  const float* gib = gi + (size_t)b * G3_;
  const float* ghb = gh + (size_t)b * G3_;
  float r = sigmoidf_(gib[j] + ghb[j]);
  float z = sigmoidf_(gib[H_ + j] + ghb[H_ + j]);
  float n = fast_tanh(gib[2 * H_ + j] + r * ghb[2 * H_ + j]);
  float hn = (1.0f - z) * n + z * h[idx];
  h[idx] = hn;
  h_bf[idx] = f2bf(hn);
}

// score[b,te] = sum_a tanh(q[b,a] + enc_proj_bf16[b,te,a](+biases folded) + conv) * v[a]
__global__ __launch_bounds__(256) void attn_score_k(
    const float* __restrict__ q, const __bf16* __restrict__ enc_proj,
    const float* __restrict__ conv_w, const float* __restrict__ attn_v,
    const float* __restrict__ attn_prev, float* __restrict__ score) {
  __shared__ float sh_cw[ATTN_ * 3];
  __shared__ float sh_v[ATTN_];
  __shared__ float sh_q[ATTN_];
  const int b = blockIdx.y;
  for (int i = threadIdx.x; i < ATTN_ * 3; i += 256) sh_cw[i] = conv_w[i];
  for (int i = threadIdx.x; i < ATTN_; i += 256) {
    sh_v[i] = attn_v[i];
    sh_q[i] = q[(size_t)b * ATTN_ + i];
  }
  __syncthreads();

  const int wave = threadIdx.x >> 5;
  const int lane = threadIdx.x & 31;
  const int te = blockIdx.x * 8 + wave;
  const float* ap = attn_prev + (size_t)b * ENC_T_;
  const float s0 = (te > 0) ? ap[te - 1] : 0.0f;
  const float s1 = ap[te];
  const float s2 = (te < ENC_T_ - 1) ? ap[te + 1] : 0.0f;
  const __bf16* ep = enc_proj + ((size_t)b * ENC_T_ + te) * ATTN_;

  float acc = 0.0f;
#pragma unroll 4
  for (int i = 0; i < ATTN_ / 32; ++i) {
    const int a = lane + 32 * i;
    float conv = sh_cw[a * 3] * s0 + sh_cw[a * 3 + 1] * s1 + sh_cw[a * 3 + 2] * s2;
    float e = fast_tanh(sh_q[a] + bf2f(ep[a]) + conv);
    acc += e * sh_v[a];
  }
  for (int off = 16; off > 0; off >>= 1) acc += __shfl_down(acc, off, 32);
  if (lane == 0) score[(size_t)b * ENC_T_ + te] = acc;
}

__global__ __launch_bounds__(256) void softmax_attn_k(
    const float* __restrict__ score, float* __restrict__ attn_w,
    float* __restrict__ aw_out, int t) {
  __shared__ float red[8];
  __shared__ float bc;
  const int b = blockIdx.x;
  const int tid = threadIdx.x;
  const float* s = score + (size_t)b * ENC_T_;

  float vals[4];
  float m = -3.0e38f;
#pragma unroll
  for (int i = 0; i < 4; ++i) { vals[i] = s[tid + i * 256]; m = fmaxf(m, vals[i]); }
  for (int off = 16; off > 0; off >>= 1) m = fmaxf(m, __shfl_xor(m, off, 32));
  if ((tid & 31) == 0) red[tid >> 5] = m;
  __syncthreads();
  if (tid == 0) {
    float x = red[0];
    for (int i = 1; i < 8; ++i) x = fmaxf(x, red[i]);
    bc = x;
  }
  __syncthreads();
  m = bc;

  float sum = 0.0f;
#pragma unroll
  for (int i = 0; i < 4; ++i) { vals[i] = expf(vals[i] - m); sum += vals[i]; }
  for (int off = 16; off > 0; off >>= 1) sum += __shfl_xor(sum, off, 32);
  if ((tid & 31) == 0) red[tid >> 5] = sum;
  __syncthreads();
  if (tid == 0) {
    float x = 0.0f;
    for (int i = 0; i < 8; ++i) x += red[i];
    bc = 1.0f / x;
  }
  __syncthreads();
  const float inv = bc;
#pragma unroll
  for (int i = 0; i < 4; ++i) {
    const int te = tid + i * 256;
    const float w = vals[i] * inv;
    attn_w[(size_t)b * ENC_T_ + te] = w;
    aw_out[((size_t)b * DEC_T_ + t) * ENC_T_ + te] = w;
  }
}

__global__ void context_k(const float* __restrict__ attn_w,
                          const float* __restrict__ enc,
                          float* __restrict__ ctx) {
  const int b = blockIdx.y;
  const int e = blockIdx.x * 256 + threadIdx.x;
  const float* aw = attn_w + (size_t)b * ENC_T_;
  const float* ep = enc + (size_t)b * ENC_T_ * ENC_D_ + e;
  float acc = 0.0f;
  for (int tt = 0; tt < ENC_T_; ++tt) {
    // speculative prefetch (global_prefetch_b8) of the line 8 rows ahead;
    // OOB prefetches are dropped silently per ISA 10.5.
    __builtin_prefetch(ep + (size_t)(tt + 8) * ENC_D_, 0, 0);
    acc += aw[tt] * ep[(size_t)tt * ENC_D_];
  }
  ctx[(size_t)b * ENC_D_ + e] = acc;
}

__global__ void pack_out_cat_k(const float* __restrict__ h,
                               const float* __restrict__ ctx,
                               float* __restrict__ oc_out,
                               __bf16* __restrict__ oc_bf, int t) {
  int idx = blockIdx.x * 256 + threadIdx.x;  // B*RIN
  int b = idx >> 10, j = idx & (RIN_ - 1);
  float v = (j < H_) ? h[b * H_ + j] : ctx[b * ENC_D_ + (j - H_)];
  oc_out[((size_t)b * DEC_T_ + t) * RIN_ + j] = v;
  oc_bf[idx] = f2bf(v);
}

__global__ __launch_bounds__(128) void log_softmax_k(
    const float* __restrict__ logits, float* __restrict__ y_out, int t) {
  __shared__ float red[4];
  __shared__ float bc;
  const int b = blockIdx.x;
  const int tid = threadIdx.x;
  const float x = logits[(size_t)b * VOCAB_ + tid];

  float m = x;
  for (int off = 16; off > 0; off >>= 1) m = fmaxf(m, __shfl_xor(m, off, 32));
  if ((tid & 31) == 0) red[tid >> 5] = m;
  __syncthreads();
  if (tid == 0) bc = fmaxf(fmaxf(red[0], red[1]), fmaxf(red[2], red[3]));
  __syncthreads();
  m = bc;

  float s = expf(x - m);
  for (int off = 16; off > 0; off >>= 1) s += __shfl_xor(s, off, 32);
  if ((tid & 31) == 0) red[tid >> 5] = s;
  __syncthreads();
  if (tid == 0) bc = m + logf(red[0] + red[1] + red[2] + red[3]);
  __syncthreads();
  y_out[((size_t)b * DEC_T_ + t) * VOCAB_ + tid] = x - bc;
}

// ---------------------------------------------------------------------------
extern "C" void kernel_launch(void* const* d_in, const int* in_sizes, int n_in,
                              void* d_out, int out_size, void* d_ws, size_t ws_size,
                              hipStream_t stream) {
  (void)in_sizes; (void)n_in; (void)out_size; (void)ws_size;
  const int*   inputs      = (const int*)  d_in[0];
  const float* enc_out     = (const float*)d_in[1];
  const float* embedding   = (const float*)d_in[2];
  const float* w_ih        = (const float*)d_in[3];
  const float* w_hh        = (const float*)d_in[4];
  const float* b_ih        = (const float*)d_in[5];
  const float* b_hh        = (const float*)d_in[6];
  const float* attn_wq     = (const float*)d_in[7];
  const float* attn_wv     = (const float*)d_in[8];
  const float* attn_conv_w = (const float*)d_in[9];
  const float* attn_conv_b = (const float*)d_in[10];
  const float* attn_bias   = (const float*)d_in[11];
  const float* attn_v      = (const float*)d_in[12];
  const float* fc_w        = (const float*)d_in[13];
  const float* fc_b        = (const float*)d_in[14];

  float* y_out  = (float*)d_out;                               // (B,T,VOCAB)
  float* aw_out = y_out + (size_t)B_ * DEC_T_ * VOCAB_;        // (B,T,ENC_T)
  float* oc_out = aw_out + (size_t)B_ * DEC_T_ * ENC_T_;       // (B,T,RIN)

  // workspace carve-up (256B aligned slabs)
  char* w = (char*)d_ws;
  auto alloc = [&](size_t bytes) {
    char* p = w;
    w += (bytes + 255) & ~(size_t)255;
    return p;
  };
  __bf16* enc_proj = (__bf16*)alloc((size_t)B_ * ENC_T_ * ATTN_ * 2); // 64 MB
  __bf16* emb_bf   = (__bf16*)alloc((size_t)B_ * DEC_T_ * H_ * 2);    //  8 MB
  __bf16* w_ih_bf  = (__bf16*)alloc((size_t)G3_ * RIN_ * 2);
  __bf16* w_hh_bf  = (__bf16*)alloc((size_t)G3_ * H_ * 2);
  __bf16* wq_bf    = (__bf16*)alloc((size_t)ATTN_ * H_ * 2);
  __bf16* wv_bf    = (__bf16*)alloc((size_t)ATTN_ * ENC_D_ * 2);
  __bf16* fc_bf    = (__bf16*)alloc((size_t)VOCAB_ * RIN_ * 2);
  float*  bias_sum = (float*) alloc((size_t)ATTN_ * 4);
  __bf16* rnn_in   = (__bf16*)alloc((size_t)B_ * RIN_ * 2);
  float*  h        = (float*) alloc((size_t)B_ * H_ * 4);
  __bf16* h_bf     = (__bf16*)alloc((size_t)B_ * H_ * 2);
  float*  ctx      = (float*) alloc((size_t)B_ * ENC_D_ * 4);
  float*  gi       = (float*) alloc((size_t)B_ * G3_ * 4);
  float*  gh       = (float*) alloc((size_t)B_ * G3_ * 4);
  float*  q        = (float*) alloc((size_t)B_ * ATTN_ * 4);
  float*  score    = (float*) alloc((size_t)B_ * ENC_T_ * 4);
  float*  attn_w   = (float*) alloc((size_t)B_ * ENC_T_ * 4);
  __bf16* oc_bf    = (__bf16*)alloc((size_t)B_ * RIN_ * 2);
  float*  logits   = (float*) alloc((size_t)B_ * VOCAB_ * 4);

  // recurrent state -> zeros (deterministic per call)
  hipMemsetAsync(h,      0, (size_t)B_ * H_ * 4, stream);
  hipMemsetAsync(h_bf,   0, (size_t)B_ * H_ * 2, stream);
  hipMemsetAsync(ctx,    0, (size_t)B_ * ENC_D_ * 4, stream);
  hipMemsetAsync(attn_w, 0, (size_t)B_ * ENC_T_ * 4, stream);

  // one-time conversions + embedding gather
  cvt_f32_bf16<<<(G3_ * RIN_) / 256, 256, 0, stream>>>(w_ih, w_ih_bf, G3_ * RIN_);
  cvt_f32_bf16<<<(G3_ * H_) / 256, 256, 0, stream>>>(w_hh, w_hh_bf, G3_ * H_);
  cvt_f32_bf16<<<(ATTN_ * H_) / 256, 256, 0, stream>>>(attn_wq, wq_bf, ATTN_ * H_);
  cvt_f32_bf16<<<(ATTN_ * ENC_D_) / 256, 256, 0, stream>>>(attn_wv, wv_bf, ATTN_ * ENC_D_);
  cvt_f32_bf16<<<(VOCAB_ * RIN_) / 256, 256, 0, stream>>>(fc_w, fc_bf, VOCAB_ * RIN_);
  bias_sum_k<<<2, 256, 0, stream>>>(attn_conv_b, attn_bias, bias_sum, ATTN_);
  embed_k<<<(B_ * DEC_T_ * H_) / 256, 256, 0, stream>>>(inputs, embedding, emb_bf);

  // enc_proj = encoder_outputs @ attn_wv^T + (conv_b + bias)   [bf16 out, L2-resident]
  gemm_bf16_wmma<true, true>
      <<<dim3(ATTN_ / 64, (B_ * ENC_T_) / 16), 32, 0, stream>>>(
          enc_out, wv_bf, bias_sum, enc_proj, B_ * ENC_T_, ATTN_, ENC_D_);

  for (int t = 0; t < DEC_T_; ++t) {
    pack_rnn_in_k<<<(B_ * RIN_) / 256, 256, 0, stream>>>(emb_bf, ctx, rnn_in, t);

    gemm_bf16_wmma<false, false>
        <<<dim3(G3_ / 64, B_ / 16), 32, 0, stream>>>(
            rnn_in, w_ih_bf, b_ih, gi, B_, G3_, RIN_);
    gemm_bf16_wmma<false, false>
        <<<dim3(G3_ / 64, B_ / 16), 32, 0, stream>>>(
            h_bf, w_hh_bf, b_hh, gh, B_, G3_, H_);

    gru_update_k<<<(B_ * H_) / 256, 256, 0, stream>>>(gi, gh, h, h_bf);

    gemm_bf16_wmma<false, false>
        <<<dim3(ATTN_ / 64, B_ / 16), 32, 0, stream>>>(
            h_bf, wq_bf, nullptr, q, B_, ATTN_, H_);

    attn_score_k<<<dim3(ENC_T_ / 8, B_), 256, 0, stream>>>(
        q, enc_proj, attn_conv_w, attn_v, attn_w, score);

    softmax_attn_k<<<B_, 256, 0, stream>>>(score, attn_w, aw_out, t);

    context_k<<<dim3(ENC_D_ / 256, B_), 256, 0, stream>>>(attn_w, enc_out, ctx);

    pack_out_cat_k<<<(B_ * RIN_) / 256, 256, 0, stream>>>(h, ctx, oc_out, oc_bf, t);

    gemm_bf16_wmma<false, false>
        <<<dim3(VOCAB_ / 64, B_ / 16), 32, 0, stream>>>(
            oc_bf, fc_bf, fc_b, logits, B_, VOCAB_, RIN_);

    log_softmax_k<<<B_, VOCAB_, 0, stream>>>(logits, y_out, t);
  }
}
